// VectorQuantizer_61031485276535
// MI455X (gfx1250) — matrix operand order repaired and checked by reading, not compiled, not run on hardware
//
#include <hip/hip_runtime.h>

typedef __attribute__((ext_vector_type(16))) __bf16 v16bf;
typedef __attribute__((ext_vector_type(8)))  __bf16 v8bf;
typedef __attribute__((ext_vector_type(8)))  float  v8f;
typedef __attribute__((ext_vector_type(4)))  unsigned int u32x4;
typedef __attribute__((ext_vector_type(8)))  int i32x8;
typedef __attribute__((ext_vector_type(4)))  int i32x4;

constexpr int EDIM  = 256;
constexpr int NCODE = 512;
constexpr int MTOT  = 32 * 16 * 16 * 16;          // 131072 rows
constexpr int ROWS_PER_BLOCK = 128;               // 8 waves * 16 rows
constexpr int NBLOCKS = MTOT / ROWS_PER_BLOCK;    // 1024

// ---- workspace layout (bytes) ----
constexpr size_t WS_EMB_BF16 = 0;                          // NCODE*EDIM bf16
constexpr size_t WS_ENORM    = (size_t)NCODE * EDIM * 2;   // 512 f32
constexpr size_t WS_LOSS     = WS_ENORM + (size_t)NCODE*4; // 1 f32

// ---- output layout (float elements, reference return order) ----
constexpr size_t OUT_ZQ   = 0;                          // 131072*256
constexpr size_t OUT_VQ   = (size_t)MTOT * EDIM;        // 33554432
constexpr size_t OUT_COMM = OUT_VQ + 1;
constexpr size_t OUT_IDX  = OUT_VQ + 2;                 // 131072 (as float)
constexpr size_t OUT_HIST = OUT_IDX + MTOT;             // 512

// dynamic LDS: bf16 codebook + enorm2 + per-wave row indices
constexpr size_t SMEM_EMB   = (size_t)NCODE * EDIM * 2;     // 262144
constexpr size_t SMEM_ENORM = SMEM_EMB + (size_t)NCODE * 4; // +2048
constexpr size_t SMEM_TOTAL = SMEM_ENORM + (size_t)ROWS_PER_BLOCK * 4; // 264704

__global__ void vq_zero(float* __restrict__ out, float* __restrict__ wloss) {
    int t = threadIdx.x + blockIdx.x * blockDim.x;
    if (t < NCODE) out[OUT_HIST + t] = 0.0f;
    if (t == 0)    *wloss = 0.0f;
}

__global__ __launch_bounds__(256)
void vq_prep(const float* __restrict__ emb, __bf16* __restrict__ wemb,
             float* __restrict__ wnorm) {
    __shared__ float red[256];
    const int n = blockIdx.x, t = threadIdx.x;
    float v = emb[(size_t)n * EDIM + t];
    wemb[(size_t)n * EDIM + t] = (__bf16)v;
    red[t] = v * v;
    __syncthreads();
    for (int s = 128; s > 0; s >>= 1) {
        if (t < s) red[t] += red[t + s];
        __syncthreads();
    }
    if (t == 0) wnorm[n] = red[0];
}

__global__ __launch_bounds__(256)
void vq_main(const float* __restrict__ z, const float* __restrict__ emb,
             const __bf16* __restrict__ wemb, const float* __restrict__ wnorm,
             float* __restrict__ out, float* __restrict__ wloss) {
    extern __shared__ unsigned char smem[];
    __bf16* eL  = (__bf16*)smem;                 // [NCODE][EDIM] bf16
    float*  enL = (float*)(smem + SMEM_EMB);     // [NCODE]
    int*    idxL = (int*)(smem + SMEM_ENORM);    // [128] per-block row winners

    const int tid = threadIdx.x;

    // ---- stage bf16 codebook into LDS ----
#if __has_builtin(__builtin_amdgcn_tensor_load_to_lds)
    // Tensor Data Mover: one wave DMAs the whole 256KB tile (512 rows x 512B),
    // tracked with TENSORcnt; waves 1-7 overlap A-fragment building with it.
    if (tid < 32) {
        const unsigned long long ga = (unsigned long long)(uintptr_t)wemb;
        const unsigned ldsOff = (unsigned)(uintptr_t)eL;   // LDS byte offset
        u32x4 g0;
        g0[0] = 1u;                                        // count=1, user D#
        g0[1] = ldsOff;                                    // lds_addr
        g0[2] = (unsigned)(ga & 0xFFFFFFFFu);              // global_addr[31:0]
        g0[3] = (unsigned)((ga >> 32) & 0x01FFFFFFu)       // global_addr[56:32]
              | (2u << 30);                                // type=2 (image)
        i32x8 g1;
        g1[0] = (3 << 16);       // wg_mask=0 (no cluster), data_size=8B
        g1[1] = (64 << 16);      // tensor_dim0 = 64  (512B row / 8B)
        g1[2] = (512 << 16);     // tensor_dim1 = 512 rows
        g1[3] = (64 << 16);      // tile_dim0 = 64
        g1[4] = 512;             // tile_dim1 = 512
        g1[5] = 64;              // tensor_dim0_stride = 64 (contiguous)
        g1[6] = 0;
        g1[7] = 0;
        i32x4 zz = {0, 0, 0, 0};
#if __clang_major__ >= 23
        i32x8 zz8 = {0, 0, 0, 0, 0, 0, 0, 0};
        __builtin_amdgcn_tensor_load_to_lds(g0, g1, zz, zz, zz8, 0);
#else
        __builtin_amdgcn_tensor_load_to_lds(g0, g1, zz, zz, 0);
#endif
    }
    for (int i = tid; i < NCODE; i += 256) enL[i] = wnorm[i];
    if (tid < 32) __builtin_amdgcn_s_wait_tensorcnt(0);
#else
    {
        const uint4* src = (const uint4*)wemb;
        uint4* dst = (uint4*)eL;
        const int nvec = (int)(SMEM_EMB / 16);   // 16384 x 16B
        for (int i = tid; i < nvec; i += 256) dst[i] = src[i];
        for (int i = tid; i < NCODE; i += 256) enL[i] = wnorm[i];
    }
#endif

    const int lane = tid & 31;
    const int wave = tid >> 5;
    const int g  = lane >> 4;        // half-wave group (K-halves)
    const int mr = lane & 15;        // row-in-tile / col-in-tile
    const int rowBase = blockIdx.x * ROWS_PER_BLOCK + wave * 16;
    const float* zrow = z + (size_t)(rowBase + mr) * EDIM;

    // ---- build bf16 A fragments (16x32 per K-step), ISA 16-bit A layout:
    // lane m holds K = [8g..8g+8) and [16+8g..16+8g+8) within each K-step ----
    v16bf A[8];
#pragma unroll
    for (int kb = 0; kb < 8; ++kb) {
        const float4* p0 = (const float4*)(zrow + kb * 32 + 8 * g);
        const float4* p1 = (const float4*)(zrow + kb * 32 + 16 + 8 * g);
        float4 q0 = p0[0], q1 = p0[1], q2 = p1[0], q3 = p1[1];
        v16bf a;
        a[0] = (__bf16)q0.x; a[1] = (__bf16)q0.y; a[2]  = (__bf16)q0.z; a[3]  = (__bf16)q0.w;
        a[4] = (__bf16)q1.x; a[5] = (__bf16)q1.y; a[6]  = (__bf16)q1.z; a[7]  = (__bf16)q1.w;
        a[8] = (__bf16)q2.x; a[9] = (__bf16)q2.y; a[10] = (__bf16)q2.z; a[11] = (__bf16)q2.w;
        a[12]= (__bf16)q3.x; a[13]= (__bf16)q3.y; a[14] = (__bf16)q3.z; a[15] = (__bf16)q3.w;
        A[kb] = a;
    }
    __syncthreads();   // codebook + norms visible to all waves

    // ---- running argmax of (||e||^2 - 2 z.e) over all 512 codes ----
    float best[8];
    int   bidx[8];
#pragma unroll
    for (int r = 0; r < 8; ++r) { best[r] = -3.4e38f; bidx[r] = 0; }

#pragma unroll 2
    for (int nt = 0; nt < NCODE / 16; ++nt) {
        const int n0 = nt * 16;
        const __bf16* bbase = eL + (size_t)(n0 + mr) * EDIM;
        v8f c = {};
#pragma unroll
        for (int kb = 0; kb < 8; ++kb) {
            // B fragment (32x16 K-major): lane n holds K=[8g..8g+8) and [16+8g..16+8g+8)
            const v8bf* bp = (const v8bf*)(bbase + kb * 32 + 8 * g);
            const v8bf* bq = (const v8bf*)(bbase + kb * 32 + 16 + 8 * g);
            v8bf b0 = bp[0], b1 = bq[0];
            v16bf b;
#pragma unroll
            for (int h = 0; h < 8; ++h) { b[h] = b0[h]; b[8 + h] = b1[h]; }
            c = __builtin_amdgcn_wmma_f32_16x16x32_bf16(
                    false, A[kb], false, b, (short)0, c, false, false);
        }
        const float en  = enL[n0 + mr];
        const int   myn = n0 + mr;
#pragma unroll
        for (int r = 0; r < 8; ++r) {
            float s = en - 2.0f * c[r];
            if (s > best[r]) { best[r] = s; bidx[r] = myn; }  // strict > keeps first max
        }
    }

    // ---- cross-lane argmax within each 16-lane group (ties -> lowest idx) ----
#pragma unroll
    for (int r = 0; r < 8; ++r) {
#pragma unroll
        for (int m = 1; m < 16; m <<= 1) {
            float os = __shfl_xor(best[r], m, 32);
            int   oi = __shfl_xor(bidx[r], m, 32);
            if (os > best[r] || (os == best[r] && oi < bidx[r])) {
                best[r] = os; bidx[r] = oi;
            }
        }
        if (mr == 0) {
            const int rowInWave = r + 8 * g;     // lane0 -> rows 0..7, lane16 -> 8..15
            const int grow = rowBase + rowInWave;
            idxL[wave * 16 + rowInWave] = bidx[r];
            out[OUT_IDX + grow] = (float)bidx[r];
            atomicAdd(&out[OUT_HIST + bidx[r]], 1.0f);
        }
    }
    __syncthreads();

    // ---- gather z_q = E[idx] (f32 from global/L2), write out, loss partials ----
    float lsum = 0.0f;
    for (int rr = 0; rr < 16; ++rr) {
        const int grow = rowBase + rr;
        const int ci = idxL[wave * 16 + rr];
        const float4* es = (const float4*)(emb + (size_t)ci * EDIM);
        const float4* zs = (const float4*)(z + (size_t)grow * EDIM);
        float4* d = (float4*)(out + OUT_ZQ + (size_t)grow * EDIM);
        for (int j = lane; j < EDIM / 4; j += 32) {
            float4 e = es[j], zv = zs[j];
            d[j] = e;
            float dx = e.x - zv.x, dy = e.y - zv.y, dz = e.z - zv.z, dw = e.w - zv.w;
            lsum += dx * dx + dy * dy + dz * dz + dw * dw;
        }
    }
#pragma unroll
    for (int m = 1; m < 32; m <<= 1) lsum += __shfl_xor(lsum, m, 32);
    if (lane == 0) atomicAdd(wloss, lsum);
}

__global__ void vq_fin(const float* __restrict__ wloss, float* __restrict__ out) {
    float vq = *wloss / (float)((size_t)MTOT * EDIM);
    out[OUT_VQ]   = vq;
    out[OUT_COMM] = 0.25f * vq;
}

extern "C" void kernel_launch(void* const* d_in, const int* in_sizes, int n_in,
                              void* d_out, int out_size, void* d_ws, size_t ws_size,
                              hipStream_t stream) {
    (void)in_sizes; (void)n_in; (void)out_size; (void)ws_size;
    const float* z   = (const float*)d_in[0];
    const float* emb = (const float*)d_in[1];
    float* out = (float*)d_out;
    unsigned char* ws = (unsigned char*)d_ws;

    __bf16* wemb  = (__bf16*)(ws + WS_EMB_BF16);
    float*  wnorm = (float*)(ws + WS_ENORM);
    float*  wloss = (float*)(ws + WS_LOSS);

    vq_zero<<<2, 256, 0, stream>>>(out, wloss);
    vq_prep<<<NCODE, 256, 0, stream>>>(emb, wemb, wnorm);

    // 256KB+ dynamic LDS: legal on CDNA5 (320KB/WGP); raise the cap.
    hipFuncSetAttribute(reinterpret_cast<const void*>(vq_main),
                        hipFuncAttributeMaxDynamicSharedMemorySize,
                        (int)SMEM_TOTAL);
    vq_main<<<NBLOCKS, 256, SMEM_TOTAL, stream>>>(z, emb, wemb, wnorm, out, wloss);
    vq_fin<<<1, 1, 0, stream>>>(wloss, out);
}
